// NonLinearOp_36558761624009
// MI455X (gfx1250) — matrix operand order repaired and checked by reading, not compiled, not run on hardware
//
#include <hip/hip_runtime.h>
#include <hip/hip_bf16.h>

// SNN integrate-and-fire (elementwise fc1 + IAF over T on VALU) with the
// 64-neuron fc2 reduction done on the CDNA5 matrix pipe via
// v_wmma_f32_16x16x32_f16. Spike indicators are exact in f16; W2*thr is
// split into f16 hi+lo parts (two chained WMMAs) for f32-level accuracy.

typedef __attribute__((ext_vector_type(16))) _Float16 v16h;
typedef __attribute__((ext_vector_type(8)))  float    v8f;

#define T_STEPS 8
#define BATCH   16
#define DIM     4096

__global__ __launch_bounds__(256, 1) void snn_iaf_wmma(
    const float* __restrict__ x,  const float* __restrict__ W1,
    const float* __restrict__ b1, const float* __restrict__ W2,
    const float* __restrict__ b2p, const float* __restrict__ thrp,
    float* __restrict__ y)
{
    const int lane = threadIdx.x & 31;
    const int wave = threadIdx.x >> 5;
    const int tile = blockIdx.x * 8 + wave;   // 4096 tiles total
    const int bidx = tile >> 8;               // batch index 0..15 (D/16 = 256 tiles per b)
    const int d0   = (tile & 255) << 4;       // 16 consecutive d columns
    const int m    = lane & 15;               // element (row M) owned by this lane
    const int g    = lane >> 4;               // lane group (selects K half per WMMA layout)

    const float thr = thrp[0];
    const float b2  = b2p[0];

    // x[t] for this lane's element: row = t*B + bidx, col = d0 + m.
    float xs[T_STEPS];
#pragma unroll
    for (int t = 0; t < T_STEPS; ++t)
        xs[t] = x[(t * BATCH + bidx) * DIM + d0 + m];

    v8f zero = {0.f, 0.f, 0.f, 0.f, 0.f, 0.f, 0.f, 0.f};
    v8f acc[T_STEPS];
#pragma unroll
    for (int t = 0; t < T_STEPS; ++t) acc[t] = zero;

    // Two 32-neuron chunks cover all 64 neurons (K=32 per WMMA).
#pragma unroll
    for (int c = 0; c < 2; ++c) {
        // A-matrix 16x32 f16 layout: this lane's 16 K-slots are
        // e=0..7 -> K = 8g+e ; e=8..15 -> K = 16+8g+(e-8).
        const int nA0 = c * 32 + 8 * g;
        const int nA1 = nA0 + 16;
        // B-matrix 32x16 layout: e=0..7 -> K = 8g+e ; e=8..15 -> K = 16+8g+(e-8).
        const int nB  = c * 32 + 8 * g;

        float w1v[16], b1v[16], vm[16];
#pragma unroll
        for (int e = 0; e < 8; ++e) {
            w1v[e]     = W1[nA0 + e];
            w1v[e + 8] = W1[nA1 + e];
            b1v[e]     = b1[nA0 + e];
            b1v[e + 8] = b1[nA1 + e];
        }
#pragma unroll
        for (int e = 0; e < 16; ++e) vm[e] = 0.f;

        // B = thr*W2 broadcast across all N columns, split into f16 hi+lo.
        v16h bh, bl;
#pragma unroll
        for (int e = 0; e < 16; ++e) {
            float w2s = __fmul_rn(thr, W2[nB + (e < 8 ? e : e + 8)]);
            _Float16 hi = (_Float16)w2s;
            bh[e] = hi;
            bl[e] = (_Float16)__fsub_rn(w2s, (float)hi);
        }

#pragma unroll
        for (int t = 0; t < T_STEPS; ++t) {
            v16h a;
#pragma unroll
            for (int e = 0; e < 16; ++e) {
                // Match reference rounding exactly (no fp-contract): the
                // v >= thr decision is discontinuous, so bit-identical v matters.
                float h = __fadd_rn(__fmul_rn(xs[t], w1v[e]), b1v[e]);
                float v = __fadd_rn(vm[e], h);
                bool fired = v >= thr;
                vm[e] = __fsub_rn(v, fired ? thr : 0.0f);
                a[e]  = fired ? (_Float16)1.0f : (_Float16)0.0f;
            }
            // D = A(ind) x B(thr*W2 hi) + C ; then + A x B(lo): f32-accurate dot.
            acc[t] = __builtin_amdgcn_wmma_f32_16x16x32_f16(
                false, a, false, bh, (short)0, acc[t], false, false);
            acc[t] = __builtin_amdgcn_wmma_f32_16x16x32_f16(
                false, a, false, bl, (short)0, acc[t], false, false);
        }
    }

    // C/D layout: VGPR r at group g holds M = r + 8g (all N columns identical).
    // Lanes 0 and 16 store M=0..7 and M=8..15 as two float4 each per timestep.
    if (m == 0) {
#pragma unroll
        for (int t = 0; t < T_STEPS; ++t) {
            const int base = (t * BATCH + bidx) * DIM + d0 + 8 * g;
            float4 o0, o1;
            o0.x = __fadd_rn(acc[t][0], b2);
            o0.y = __fadd_rn(acc[t][1], b2);
            o0.z = __fadd_rn(acc[t][2], b2);
            o0.w = __fadd_rn(acc[t][3], b2);
            o1.x = __fadd_rn(acc[t][4], b2);
            o1.y = __fadd_rn(acc[t][5], b2);
            o1.z = __fadd_rn(acc[t][6], b2);
            o1.w = __fadd_rn(acc[t][7], b2);
            *reinterpret_cast<float4*>(y + base)     = o0;
            *reinterpret_cast<float4*>(y + base + 4) = o1;
        }
    }
}

extern "C" void kernel_launch(void* const* d_in, const int* in_sizes, int n_in,
                              void* d_out, int out_size, void* d_ws, size_t ws_size,
                              hipStream_t stream) {
    const float* x   = (const float*)d_in[0];
    const float* W1  = (const float*)d_in[1];
    const float* b1  = (const float*)d_in[2];
    const float* W2  = (const float*)d_in[3];
    const float* b2  = (const float*)d_in[4];
    const float* thr = (const float*)d_in[5];
    // d_in[6] is T (== 8), baked in as a compile-time constant.
    float* y = (float*)d_out;

    // 16*4096 (b,d) pairs / 16 per wave = 4096 waves; 8 waves per block.
    dim3 grid(512), block(256);
    hipLaunchKernelGGL(snn_iaf_wmma, grid, block, 0, stream,
                       x, W1, b1, W2, b2, thr, y);
}
